// SignalOptimizer_30159260352555
// MI455X (gfx1250) — compile-verified, matrix operand order checked by reading, not run on hardware
//
#include <hip/hip_runtime.h>
#include <hip/hip_bf16.h>
#include <cstdint>

// ---------------------------------------------------------------------------
// CDNA5 WMMA helpers (wave32, v_wmma_f32_16x16x32_f16)
// ---------------------------------------------------------------------------
typedef __attribute__((ext_vector_type(16))) _Float16 v16h;
typedef __attribute__((ext_vector_type(8)))  _Float16 v8h;
typedef __attribute__((ext_vector_type(8)))  float    v8f;

union Frag16 { v16h f; v8h h8[2]; };

// A fragment (16x32, row r = lane&15) and B fragment (32x16 = W^T with
// n = lane&15 reading rows of W[N x K]) share the same lane->K mapping:
// per lane two contiguous 8-half (16B) runs at k0 + 8*(lane>=16) and +16.
__device__ __forceinline__ v16h load_frag(const _Float16* __restrict__ base,
                                          int row0, int ld, int k0, int lane) {
  const _Float16* p = base + (size_t)(row0 + (lane & 15)) * (size_t)ld
                           + k0 + ((lane >> 4) << 3);
  Frag16 fr;
  fr.h8[0] = *(const v8h*)(p);
  fr.h8[1] = *(const v8h*)(p + 16);
  return fr.f;
}

__device__ __forceinline__ v8f wmma_f16(v16h a, v16h b, v8f c) {
  return __builtin_amdgcn_wmma_f32_16x16x32_f16(false, a, false, b,
                                                (short)0, c, false, false);
}

__device__ __forceinline__ float sigmoidf_(float x) { return 1.f / (1.f + __expf(-x)); }

// ---------------------------------------------------------------------------
// Problem constants
// ---------------------------------------------------------------------------
static constexpr int B_   = 512;
static constexpr int T_   = 32;       // Tp + Tf
static constexpr int Dd_  = 64;
static constexpr int N_   = 8;
static constexpr int D_   = 64;
static constexpr int RH_  = 128;
static constexpr int BT_  = B_ * T_;          // 16384
static constexpr int M_GNN = BT_ * N_;        // 131072

// ---------------------------------------------------------------------------
// Weight conversion kernels
// ---------------------------------------------------------------------------
__global__ void f32_to_f16_kernel(const float* __restrict__ src,
                                  _Float16* __restrict__ dst, int n) {
  int i = blockIdx.x * blockDim.x + threadIdx.x;
  if (i < n) dst[i] = (_Float16)src[i];
}

// msg_W (D x 2D) -> packed Wcat (2D x D) f16: rows 0..63 = mW[:, :64],
// rows 64..127 = mW[:, 64:], so pre = h @ Wcat^T gives [pre_i | pre_j].
__global__ void pack_msgW_kernel(const float* __restrict__ mW,
                                 _Float16* __restrict__ dst) {
  int i = blockIdx.x * blockDim.x + threadIdx.x;   // 128*64
  if (i >= 128 * 64) return;
  int o = i >> 6, k = i & 63;
  float v = (o < 64) ? mW[o * 128 + k] : mW[(o - 64) * 128 + 64 + k];
  dst[i] = (_Float16)v;
}

// ---------------------------------------------------------------------------
// Generic WMMA GEMM: C[M x N] = A[M x K] * W[N x K]^T   (f16 in, f32 acc)
// wave-per-16x16 tile; all dims multiples of 16/32 -> EXEC all-ones at WMMA.
// ---------------------------------------------------------------------------
template <bool HALF_OUT>
__global__ void gemm_wmma_kernel(const _Float16* __restrict__ A,
                                 const _Float16* __restrict__ W,
                                 float* __restrict__ Cf,
                                 _Float16* __restrict__ Ch,
                                 int M, int K, int N) {
  int lane = threadIdx.x & 31;
  int wid  = blockIdx.x * (blockDim.x >> 5) + (threadIdx.x >> 5);
  int tilesN = N >> 4;
  int tiles  = (M >> 4) * tilesN;
  if (wid >= tiles) return;
  int m0 = (wid / tilesN) << 4;
  int n0 = (wid % tilesN) << 4;
  v8f acc = {};
  for (int k0 = 0; k0 < K; k0 += 32) {
    if (k0 + 32 < K)   // prefetch next A k-block (global_prefetch_b8 path)
      __builtin_prefetch(A + (size_t)(m0 + (lane & 15)) * K + k0 + 32, 0, 1);
    v16h a = load_frag(A, m0, K, k0, lane);
    v16h b = load_frag(W, n0, K, k0, lane);
    acc = wmma_f16(a, b, acc);
  }
  int n  = n0 + (lane & 15);
  int mb = m0 + ((lane >> 4) << 3);
  if (HALF_OUT) {
#pragma unroll
    for (int v = 0; v < 8; ++v) Ch[(size_t)(mb + v) * N + n] = (_Float16)acc[v];
  } else {
#pragma unroll
    for (int v = 0; v < 8; ++v) Cf[(size_t)(mb + v) * N + n] = acc[v];
  }
}

// ---------------------------------------------------------------------------
// Stage 1: lane embedding  h[m,d] = (x_row . dl_W[n]) + dl_b[n]) * embW[d] + embb[d]
// ---------------------------------------------------------------------------
__global__ void embed_kernel(const float* __restrict__ past,
                             const float* __restrict__ fut,
                             const float* __restrict__ dlW,
                             const float* __restrict__ dlb,
                             const float* __restrict__ embW,
                             const float* __restrict__ embb,
                             float* __restrict__ h32,
                             _Float16* __restrict__ h16) {
  int m = blockIdx.x * blockDim.x + threadIdx.x;   // M_GNN
  if (m >= M_GNN) return;
  int bt = m >> 3, nd = m & 7;
  int b = bt >> 5, t = bt & 31;
  const float* x = (t < 16) ? past + ((size_t)b * 16 + t) * Dd_
                            : fut  + ((size_t)b * 16 + (t - 16)) * Dd_;
  const float* w = dlW + nd * Dd_;
  float lane = dlb[nd];
#pragma unroll 8
  for (int k = 0; k < Dd_; ++k) lane += x[k] * w[k];
  float*    ho = h32 + (size_t)m * D_;
  _Float16* hh = h16 + (size_t)m * D_;
#pragma unroll 8
  for (int d = 0; d < D_; ++d) {
    float v = lane * embW[d] + embb[d];
    ho[d] = v; hh[d] = (_Float16)v;
  }
}

// ---------------------------------------------------------------------------
// Message aggregation: agg[bt,i,d] = sum_{j!=i} relu(pre_i[bt,i,d]+pre_j[bt,j,d]+mb[d])
// pre packed (M_GNN x 128): cols 0:64 = pre_i, 64:128 = pre_j
// ---------------------------------------------------------------------------
__global__ void msg_agg_kernel(const _Float16* __restrict__ pre,
                               const float* __restrict__ mb,
                               _Float16* __restrict__ agg) {
  int idx = blockIdx.x * blockDim.x + threadIdx.x;   // M_GNN * 64
  if (idx >= M_GNN * D_) return;
  int d = idx & 63;
  int m = idx >> 6;
  int bt = m >> 3, i = m & 7;
  float pi = (float)pre[(size_t)m * 128 + d] + mb[d];
  const _Float16* pj = pre + ((size_t)bt * 8) * 128 + 64 + d;
  float s = 0.f;
#pragma unroll
  for (int j = 0; j < 8; ++j) {
    if (j == i) continue;
    float v = pi + (float)pj[(size_t)j * 128];
    s += v > 0.f ? v : 0.f;
  }
  agg[(size_t)m * D_ + d] = (_Float16)s;
}

// ---------------------------------------------------------------------------
// Fused GNN GRU cell: block owns 16 rows; WMMA gi (agg.Wih^T) and gh (h.Whh^T)
// into LDS, barrier, then gate math; h updated in place (rows block-exclusive).
// ---------------------------------------------------------------------------
__global__ void gnn_gru_kernel(const _Float16* __restrict__ agg16,
                               float* __restrict__ h32,
                               _Float16* __restrict__ h16,
                               const _Float16* __restrict__ Wih,
                               const _Float16* __restrict__ Whh,
                               const float* __restrict__ bih,
                               const float* __restrict__ bhh) {
  __shared__ float gi_s[16 * 192];
  __shared__ float gh_s[16 * 192];
  int m0   = blockIdx.x << 4;
  int wid  = threadIdx.x >> 5;
  int lane = threadIdx.x & 31;
  // 24 tiles (12 gi + 12 gh), 8 waves x 3 tiles, K=64 -> 2 WMMA each
  for (int tt = wid * 3; tt < wid * 3 + 3; ++tt) {
    bool isGh = tt >= 12;
    int n0 = (isGh ? tt - 12 : tt) << 4;
    const _Float16* Am = isGh ? h16 : agg16;
    const _Float16* Wm = isGh ? Whh : Wih;
    v8f acc = {};
    for (int k0 = 0; k0 < 64; k0 += 32) {
      v16h a = load_frag(Am, m0, 64, k0, lane);
      v16h b = load_frag(Wm, n0, 64, k0, lane);
      acc = wmma_f16(a, b, acc);
    }
    float* dst = isGh ? gh_s : gi_s;
    int n  = n0 + (lane & 15);
    int mb = (lane >> 4) << 3;
#pragma unroll
    for (int v = 0; v < 8; ++v) dst[(mb + v) * 192 + n] = acc[v];
  }
  __syncthreads();
  for (int e = threadIdx.x; e < 16 * 64; e += 256) {
    int r = e >> 6, d = e & 63;
    size_t m = (size_t)(m0 + r);
    float gir = gi_s[r * 192 + d]       + bih[d];
    float giz = gi_s[r * 192 + 64 + d]  + bih[64 + d];
    float gin = gi_s[r * 192 + 128 + d] + bih[128 + d];
    float ghr = gh_s[r * 192 + d]       + bhh[d];
    float ghz = gh_s[r * 192 + 64 + d]  + bhh[64 + d];
    float ghn = gh_s[r * 192 + 128 + d] + bhh[128 + d];
    float rg = sigmoidf_(gir + ghr);
    float z  = sigmoidf_(giz + ghz);
    float nn = tanhf(gin + rg * ghn);
    float hold = h32[m * D_ + d];
    float hn = (1.f - z) * nn + z * hold;
    h32[m * D_ + d] = hn;
    h16[m * D_ + d] = (_Float16)hn;
  }
}

// ---------------------------------------------------------------------------
// LayerNorm over D=64, one wave per row (wave32: 2 elems/lane, shfl_xor reduce)
// ---------------------------------------------------------------------------
__global__ void layernorm_kernel(float* __restrict__ h32, _Float16* __restrict__ h16,
                                 const float* __restrict__ g, const float* __restrict__ b) {
  int row  = blockIdx.x * (blockDim.x >> 5) + (threadIdx.x >> 5);
  int lane = threadIdx.x & 31;
  float* p = h32 + (size_t)row * D_;
  float x0 = p[lane], x1 = p[lane + 32];
  float s = x0 + x1;
#pragma unroll
  for (int o = 16; o > 0; o >>= 1) s += __shfl_xor(s, o, 32);
  float mu = s * (1.f / 64.f);
  float d0 = x0 - mu, d1 = x1 - mu;
  float v = d0 * d0 + d1 * d1;
#pragma unroll
  for (int o = 16; o > 0; o >>= 1) v += __shfl_xor(v, o, 32);
  float inv = rsqrtf(v * (1.f / 64.f) + 1e-5f);
  float y0 = d0 * inv * g[lane]      + b[lane];
  float y1 = d1 * inv * g[lane + 32] + b[lane + 32];
  p[lane] = y0; p[lane + 32] = y1;
  _Float16* q = h16 + (size_t)row * D_;
  q[lane] = (_Float16)y0; q[lane + 32] = (_Float16)y1;
}

// ---------------------------------------------------------------------------
// Zero the recurrent state
// ---------------------------------------------------------------------------
__global__ void zero_state_kernel(float* __restrict__ h32, _Float16* __restrict__ h16, int n) {
  int i = blockIdx.x * blockDim.x + threadIdx.x;
  if (i < n) { h32[i] = 0.f; h16[i] = (_Float16)0.f; }
}

// ---------------------------------------------------------------------------
// Fused temporal GRU step: gi precomputed (M=B*T x 384); gh via WMMA from the
// f16 hidden state (K=128) into LDS; gate math updates h in place and
// optionally emits y (f16) for the next layer's gi GEMM.
// ---------------------------------------------------------------------------
__global__ void rnn_step_kernel(const float* __restrict__ gi,
                                float* __restrict__ h32,
                                _Float16* __restrict__ h16,
                                _Float16* __restrict__ y16,   // may be null
                                const _Float16* __restrict__ Whh,
                                const float* __restrict__ bih,
                                const float* __restrict__ bhh,
                                int t) {
  __shared__ float gh_s[16 * 384];
  int b0   = blockIdx.x << 4;            // batch rows (512 total, 32 blocks)
  int wid  = threadIdx.x >> 5;
  int lane = threadIdx.x & 31;
  for (int tt = wid * 3; tt < wid * 3 + 3; ++tt) {   // 24 tiles of 384
    int n0 = tt << 4;
    v8f acc = {};
    for (int k0 = 0; k0 < RH_; k0 += 32) {
      v16h a = load_frag(h16, b0, RH_, k0, lane);
      v16h b = load_frag(Whh, n0, RH_, k0, lane);
      acc = wmma_f16(a, b, acc);
    }
    int n  = n0 + (lane & 15);
    int mb = (lane >> 4) << 3;
#pragma unroll
    for (int v = 0; v < 8; ++v) gh_s[(mb + v) * 384 + n] = acc[v];
  }
  __syncthreads();
  for (int e = threadIdx.x; e < 16 * RH_; e += 256) {
    int r = e >> 7, d = e & 127;
    int bb = b0 + r;
    const float* g = gi + ((size_t)bb * T_ + t) * 384;
    float gir = g[d]       + bih[d];
    float giz = g[128 + d] + bih[128 + d];
    float gin = g[256 + d] + bih[256 + d];
    float ghr = gh_s[r * 384 + d]       + bhh[d];
    float ghz = gh_s[r * 384 + 128 + d] + bhh[128 + d];
    float ghn = gh_s[r * 384 + 256 + d] + bhh[256 + d];
    float rg = sigmoidf_(gir + ghr);
    float z  = sigmoidf_(giz + ghz);
    float nn = tanhf(gin + rg * ghn);
    float hold = h32[(size_t)bb * RH_ + d];
    float hn = (1.f - z) * nn + z * hold;
    h32[(size_t)bb * RH_ + d] = hn;
    h16[(size_t)bb * RH_ + d] = (_Float16)hn;
    if (y16) y16[((size_t)bb * T_ + t) * RH_ + d] = (_Float16)hn;
  }
}

// ---------------------------------------------------------------------------
// Heads: phase logits + softplus durations (tiny; scalar f32 dots)
// ---------------------------------------------------------------------------
__global__ void heads_kernel(const float* __restrict__ h,
                             const float* __restrict__ pW, const float* __restrict__ pb,
                             const float* __restrict__ dW, const float* __restrict__ db,
                             float* __restrict__ out) {
  int i = blockIdx.x * blockDim.x + threadIdx.x;   // 512*4
  if (i >= B_ * 4) return;
  int b = i >> 2, p = i & 3;
  const float* hr = h + (size_t)b * RH_;
  float accp = pb[p], accd = db[p];
#pragma unroll 8
  for (int k = 0; k < RH_; ++k) {
    float hv = hr[k];
    accp += hv * pW[p * RH_ + k];
    accd += hv * dW[p * RH_ + k];
  }
  out[b * 4 + p] = accp;
  float sp = (accd > 20.f) ? accd : log1pf(__expf(accd));
  out[B_ * 4 + b * 4 + p] = sp + 5.f;
}

// ---------------------------------------------------------------------------
// Host orchestration
// ---------------------------------------------------------------------------
extern "C" void kernel_launch(void* const* d_in, const int* in_sizes, int n_in,
                              void* d_out, int out_size, void* d_ws, size_t ws_size,
                              hipStream_t stream) {
  (void)in_sizes; (void)n_in; (void)out_size; (void)ws_size;
  const float* past = (const float*)d_in[0];
  const float* fut  = (const float*)d_in[1];
  const float* dlW  = (const float*)d_in[2];
  const float* dlb  = (const float*)d_in[3];
  const float* embW = (const float*)d_in[4];
  const float* embb = (const float*)d_in[5];
  const float* msgW[2]  = {(const float*)d_in[6],  (const float*)d_in[8]};
  const float* msgb[2]  = {(const float*)d_in[7],  (const float*)d_in[9]};
  const float* cWih[2]  = {(const float*)d_in[10], (const float*)d_in[14]};
  const float* cWhh[2]  = {(const float*)d_in[11], (const float*)d_in[15]};
  const float* cbih[2]  = {(const float*)d_in[12], (const float*)d_in[16]};
  const float* cbhh[2]  = {(const float*)d_in[13], (const float*)d_in[17]};
  const float* lng  = (const float*)d_in[18];
  const float* lnb  = (const float*)d_in[19];
  const float* rWih0 = (const float*)d_in[20];
  const float* rWhh0 = (const float*)d_in[21];
  const float* rbih0 = (const float*)d_in[22];
  const float* rbhh0 = (const float*)d_in[23];
  const float* rWih1 = (const float*)d_in[24];
  const float* rWhh1 = (const float*)d_in[25];
  const float* rbih1 = (const float*)d_in[26];
  const float* rbhh1 = (const float*)d_in[27];
  const float* phW  = (const float*)d_in[28];
  const float* phb  = (const float*)d_in[29];
  const float* duW  = (const float*)d_in[30];
  const float* dub  = (const float*)d_in[31];
  float* out = (float*)d_out;

  // ---- workspace layout (bump allocator, 256B aligned) ----
  char* ws = (char*)d_ws; size_t off = 0;
  auto alloc = [&](size_t bytes) -> char* {
    char* p = ws + off; off += (bytes + 255) & ~(size_t)255; return p;
  };
  float*    hf32  = (float*)   alloc((size_t)M_GNN * D_ * 4);
  _Float16* hf16  = (_Float16*)alloc((size_t)M_GNN * D_ * 2);
  _Float16* pre16 = (_Float16*)alloc((size_t)M_GNN * 128 * 2);
  _Float16* agg16 = (_Float16*)alloc((size_t)M_GNN * D_ * 2);
  float*    gi    = (float*)   alloc((size_t)BT_ * 384 * 4);   // reused by both RNN layers
  _Float16* y016  = (_Float16*)alloc((size_t)BT_ * RH_ * 2);
  float*    hs32  = (float*)   alloc((size_t)B_ * RH_ * 4);
  _Float16* hs16  = (_Float16*)alloc((size_t)B_ * RH_ * 2);
  _Float16* wcat[2]   = {(_Float16*)alloc(128 * 64 * 2), (_Float16*)alloc(128 * 64 * 2)};
  _Float16* cWih16[2] = {(_Float16*)alloc(192 * 64 * 2), (_Float16*)alloc(192 * 64 * 2)};
  _Float16* cWhh16[2] = {(_Float16*)alloc(192 * 64 * 2), (_Float16*)alloc(192 * 64 * 2)};
  _Float16* rWih016 = (_Float16*)alloc(384 * 512 * 2);
  _Float16* rWhh016 = (_Float16*)alloc(384 * 128 * 2);
  _Float16* rWih116 = (_Float16*)alloc(384 * 128 * 2);
  _Float16* rWhh116 = (_Float16*)alloc(384 * 128 * 2);

  auto cvt = [&](const float* s, _Float16* d, int n) {
    f32_to_f16_kernel<<<(n + 255) / 256, 256, 0, stream>>>(s, d, n);
  };

  // ---- weight conversion (cheap, once per launch) ----
  pack_msgW_kernel<<<(128 * 64 + 255) / 256, 256, 0, stream>>>(msgW[0], wcat[0]);
  pack_msgW_kernel<<<(128 * 64 + 255) / 256, 256, 0, stream>>>(msgW[1], wcat[1]);
  cvt(cWih[0], cWih16[0], 192 * 64);
  cvt(cWhh[0], cWhh16[0], 192 * 64);
  cvt(cWih[1], cWih16[1], 192 * 64);
  cvt(cWhh[1], cWhh16[1], 192 * 64);
  cvt(rWih0, rWih016, 384 * 512);
  cvt(rWhh0, rWhh016, 384 * 128);
  cvt(rWih1, rWih116, 384 * 128);
  cvt(rWhh1, rWhh116, 384 * 128);

  // ---- stage 1: embedding ----
  embed_kernel<<<M_GNN / 256, 256, 0, stream>>>(past, fut, dlW, dlb, embW, embb, hf32, hf16);

  // ---- stage 2: two GNN layers ----
  for (int l = 0; l < 2; ++l) {
    // pre = h @ Wcat^T : M=131072, K=64, N=128 -> 65536 tiles / 8 waves
    gemm_wmma_kernel<true><<<(M_GNN / 16) * (128 / 16) / 8, 256, 0, stream>>>(
        hf16, wcat[l], nullptr, pre16, M_GNN, 64, 128);
    msg_agg_kernel<<<(M_GNN * D_) / 256, 256, 0, stream>>>(pre16, msgb[l], agg16);
    gnn_gru_kernel<<<M_GNN / 16, 256, 0, stream>>>(
        agg16, hf32, hf16, cWih16[l], cWhh16[l], cbih[l], cbhh[l]);
  }

  // ---- stage 3: LayerNorm (also refreshes f16 copy = RNN input seq) ----
  layernorm_kernel<<<M_GNN / 8, 256, 0, stream>>>(hf32, hf16, lng, lnb);

  // ---- stage 4: temporal GRU layer 0 ----
  // gi0 = seq @ Wih0^T : M=16384, K=512, N=384 -> 24576 tiles
  gemm_wmma_kernel<false><<<(BT_ / 16) * (384 / 16) / 8, 256, 0, stream>>>(
      hf16, rWih016, gi, nullptr, BT_, 512, 384);
  zero_state_kernel<<<(B_ * RH_) / 256, 256, 0, stream>>>(hs32, hs16, B_ * RH_);
  for (int t = 0; t < T_; ++t)
    rnn_step_kernel<<<B_ / 16, 256, 0, stream>>>(gi, hs32, hs16, y016,
                                                 rWhh016, rbih0, rbhh0, t);

  // ---- stage 5: temporal GRU layer 1 ----
  gemm_wmma_kernel<false><<<(BT_ / 16) * (384 / 16) / 8, 256, 0, stream>>>(
      y016, rWih116, gi, nullptr, BT_, 128, 384);
  zero_state_kernel<<<(B_ * RH_) / 256, 256, 0, stream>>>(hs32, hs16, B_ * RH_);
  for (int t = 0; t < T_; ++t)
    rnn_step_kernel<<<B_ / 16, 256, 0, stream>>>(gi, hs32, hs16, nullptr,
                                                 rWhh116, rbih1, rbhh1, t);

  // ---- stage 6: heads ----
  heads_kernel<<<(B_ * 4 + 255) / 256, 256, 0, stream>>>(hs32, phW, phb, duW, dub, out);
}
